// BatchGRU_22368189678081
// MI455X (gfx1250) — matrix operand order repaired and checked by reading, not compiled, not run on hardware
//
#include <hip/hip_runtime.h>
#include <hip/hip_bf16.h>
#include <math.h>

// ---------------------------------------------------------------------------
// BatchGRU for MI455X (gfx1250): bf16 WMMA GEMMs + persistent recurrent kernel
// ---------------------------------------------------------------------------

#define BB   256   // batch (graphs)
#define HH   512   // hidden
#define TT   64    // max nodes / timesteps
#define G3   1536  // 3*H

typedef __attribute__((ext_vector_type(16))) __bf16       v16bf;
typedef __attribute__((ext_vector_type(8)))  float        v8f;
typedef __attribute__((ext_vector_type(4)))  unsigned int u32x4;

union FragU { u32x4 u[2]; v16bf v; };

__device__ __forceinline__ v16bf make_frag(u32x4 a, u32x4 b) {
  FragU t; t.u[0] = a; t.u[1] = b; return t.v;
}

// ---------------- prep kernels ----------------

__global__ void k_zero(u32x4* pf32, u32x4* pb16) {
  int i = blockIdx.x * blockDim.x + threadIdx.x;
  u32x4 z = {0u, 0u, 0u, 0u};
  if (i < (BB * TT * HH * 4) / 16) pf32[i] = z;   // padded f32: 33.5 MB
  if (i < (BB * TT * HH * 2) / 16) pb16[i] = z;   // padded bf16: 16.8 MB
}

__global__ void k_starts(const int* __restrict__ ns, int* __restrict__ starts) {
  int b = threadIdx.x;
  int s = 0;
  for (int j = 0; j < b; ++j) s += ns[j];
  starts[b] = s;
}

__global__ void k_cvt_w(const float* __restrict__ wih, const float* __restrict__ whh,
                        __bf16* __restrict__ wihb, __bf16* __restrict__ whhb) {
  int i = blockIdx.x * blockDim.x + threadIdx.x;    // 2*1536*512 = 1,572,864
  if (i < 2 * G3 * HH) {
    wihb[i] = (__bf16)wih[i];
    whhb[i] = (__bf16)whh[i];
  }
}

__global__ void k_scatter(const float* __restrict__ atom, const int* __restrict__ ns,
                          const int* __restrict__ starts,
                          float* __restrict__ pf32, __bf16* __restrict__ pb16) {
  int blk = blockIdx.x;                 // B*T blocks
  int b = blk >> 6, p = blk & 63;
  if (p >= ns[b]) return;
  size_t src = ((size_t)starts[b] + p) * HH;
  size_t dst = ((size_t)b * TT + p) * HH;
  for (int i = threadIdx.x; i < HH; i += blockDim.x) {
    float v = atom[src + i];
    pf32[dst + i] = v;
    pb16[dst + i] = (__bf16)v;
  }
}

__global__ void k_h0(const float* __restrict__ pf32, float* __restrict__ h0) {
  int i = blockIdx.x * blockDim.x + threadIdx.x;    // B*H
  if (i >= BB * HH) return;
  int b = i >> 9, c = i & (HH - 1);
  const float* p = pf32 + (size_t)b * TT * HH + c;
  float m = p[0];
  for (int t = 1; t < TT; ++t) m = fmaxf(m, p[(size_t)t * HH]);
  h0[i] = m;
}

// ---------------- input projection: xw = X @ Wih^T + b_ih ----------------
// M = 2*B*T = 32768 rows, N = 3H = 1536, K = H = 512. One 16x16 tile per wave.

__global__ void __launch_bounds__(256, 1)
k_xproj(const __bf16* __restrict__ padded_b, const __bf16* __restrict__ wihb,
        const float* __restrict__ b_ih, float* __restrict__ xw) {
  int wave = threadIdx.x >> 5, lane = threadIdx.x & 31;
  int tile  = blockIdx.x * 8 + wave;        // 196,608 tiles total
  int mtile = tile / 96;
  int ntile = tile - mtile * 96;
  int m0   = mtile << 4;
  int dir  = m0 >> 14;                      // 16384 rows per direction
  int mrow = m0 & 16383;                    // row within direction = b*64 + tau
  int r = lane & 15, half = lane >> 4;

  // A: rows of padded bf16; per-lane: two contiguous 16B chunks per k-step
  const __bf16* ap = padded_b + (size_t)(mrow + r) * HH + half * 8;
  // B: rows of Wih (row n holds k-contiguous weights) -> two 16B chunks/lane, same k order as A
  const __bf16* bp = wihb + (size_t)dir * G3 * HH
                   + (size_t)(ntile * 16 + r) * HH + half * 8;

  v8f acc = {0.f, 0.f, 0.f, 0.f, 0.f, 0.f, 0.f, 0.f};
  for (int k0 = 0; k0 < HH; k0 += 32) {
    v16bf a = make_frag(*(const u32x4*)(ap + k0), *(const u32x4*)(ap + k0 + 16));
    v16bf b = make_frag(*(const u32x4*)(bp + k0), *(const u32x4*)(bp + k0 + 16));
    acc = __builtin_amdgcn_wmma_f32_16x16x32_bf16(false, a, false, b,
                                                  (short)0, acc, false, false);
  }

  int n = ntile * 16 + r;
  float bias = b_ih[dir * G3 + n];
  size_t obase = (size_t)(dir * 16384 + mrow) * G3 + n;
#pragma unroll
  for (int i = 0; i < 8; ++i) {
    int M = i + half * 8;                   // D layout: lane<16 -> M=i, else M=i+8
    xw[obase + (size_t)M * G3] = acc[i] + bias;
  }
}

// ---------------- persistent recurrent kernel ----------------
// 32 workgroups: blocks 0..15 = forward dir over 16 batch tiles, 16..31 = backward.
// Each WG owns 16 batch rows; h is bf16 in LDS (WMMA A) + fp32 in registers.
// Each of 8 waves owns 4 h-column tiles and their 3 gate tiles (12 reg-resident
// WMMA accumulators), so gates are evaluated straight from the WMMA D layout.

__global__ void __launch_bounds__(256, 1)
k_gru(const float* __restrict__ xw, const float* __restrict__ h0,
      const __bf16* __restrict__ whhb, const float* __restrict__ b_hh,
      const int* __restrict__ ns, float* __restrict__ out) {
  __shared__ __bf16 lds_h[16][520];   // 520 stride: 16B-aligned + bank-spread
  __shared__ int    snb[16];

  int dir   = blockIdx.x >> 4;
  int mtile = blockIdx.x & 15;
  int wave = threadIdx.x >> 5, lane = threadIdx.x & 31;
  int r = lane & 15, half = lane >> 4;

  if (threadIdx.x < 16) snb[threadIdx.x] = ns[mtile * 16 + threadIdx.x];

  const __bf16* wq = whhb + (size_t)dir * G3 * HH;
  const float*  bh = b_hh + dir * G3;

  float hreg[4][8], sreg[4][8];
#pragma unroll
  for (int j = 0; j < 4; ++j) {
    int c = (wave * 4 + j) * 16 + r;
#pragma unroll
    for (int i = 0; i < 8; ++i) {
      int row = i + half * 8;
      int b   = mtile * 16 + row;
      float v = h0[(size_t)b * HH + c];
      hreg[j][i] = v;
      sreg[j][i] = 0.f;
      lds_h[row][c] = (__bf16)v;
    }
  }
  __syncthreads();

  for (int t = 0; t < TT; ++t) {
    int tau = dir ? (TT - 1 - t) : t;

    // hint next xw row into cache while we do the GEMM
    {
      size_t xb = ((size_t)(dir * BB + mtile * 16 + (half * 8)) * TT + tau) * G3;
      __builtin_prefetch(&xw[xb + (wave * 4) * 16 + r], 0, 0);
    }

    v8f acc[4][3];
    v8f vz = {0.f, 0.f, 0.f, 0.f, 0.f, 0.f, 0.f, 0.f};
#pragma unroll
    for (int j = 0; j < 4; ++j)
#pragma unroll
      for (int g = 0; g < 3; ++g) acc[j][g] = vz;

    // gh = h @ Whh^T   (A from LDS, B from global/L2-resident Whh bf16)
    for (int k0 = 0; k0 < HH; k0 += 32) {
      v16bf a = make_frag(*(const u32x4*)&lds_h[r][k0 + half * 8],
                          *(const u32x4*)&lds_h[r][k0 + 16 + half * 8]);
#pragma unroll
      for (int j = 0; j < 4; ++j) {
#pragma unroll
        for (int g = 0; g < 3; ++g) {
          int n = ((wave * 4 + j) + 32 * g) * 16 + r;
          const __bf16* bp = wq + (size_t)n * HH + k0 + half * 8;
          v16bf b = make_frag(*(const u32x4*)bp, *(const u32x4*)(bp + 16));
          acc[j][g] = __builtin_amdgcn_wmma_f32_16x16x32_bf16(
              false, a, false, b, (short)0, acc[j][g], false, false);
        }
      }
    }
    __syncthreads();   // all A-reads of lds_h done before overwrite

    // gates + state update, straight from WMMA D layout (M=i+8*half, N=lane&15)
#pragma unroll
    for (int j = 0; j < 4; ++j) {
      int c = (wave * 4 + j) * 16 + r;
#pragma unroll
      for (int i = 0; i < 8; ++i) {
        int row = i + half * 8;
        int b   = mtile * 16 + row;
        size_t xb = ((size_t)(dir * BB + b) * TT + tau) * G3;
        float gr = acc[j][0][i] + bh[c];
        float gz = acc[j][1][i] + bh[512 + c];
        float gn = acc[j][2][i] + bh[1024 + c];
        float xr = xw[xb + c];
        float xz = xw[xb + 512 + c];
        float xn = xw[xb + 1024 + c];
        float rg = 1.f / (1.f + __expf(-(xr + gr)));
        float zg = 1.f / (1.f + __expf(-(xz + gz)));
        float ng = tanhf(xn + rg * gn);
        float hn = (1.f - zg) * ng + zg * hreg[j][i];
        hreg[j][i] = hn;
        float msk = (tau < snb[row]) ? 1.f : 0.f;
        sreg[j][i] += msk * hn;
        lds_h[row][c] = (__bf16)hn;
      }
    }
    __syncthreads();   // new h visible before next step's GEMM
  }

  // masked mean -> out[b, dir*512 + c]
#pragma unroll
  for (int j = 0; j < 4; ++j) {
    int c = (wave * 4 + j) * 16 + r;
#pragma unroll
    for (int i = 0; i < 8; ++i) {
      int row = i + half * 8;
      int b   = mtile * 16 + row;
      float inv = 1.f / (float)snb[row];
      out[(size_t)b * 1024 + dir * 512 + c] = sreg[j][i] * inv;
    }
  }
}

// ---------------- launch ----------------

extern "C" void kernel_launch(void* const* d_in, const int* in_sizes, int n_in,
                              void* d_out, int out_size, void* d_ws, size_t ws_size,
                              hipStream_t stream) {
  (void)in_sizes; (void)n_in; (void)out_size; (void)ws_size;
  const float* atom = (const float*)d_in[0];
  const float* w_ih = (const float*)d_in[1];
  const float* w_hh = (const float*)d_in[2];
  const float* b_ih = (const float*)d_in[3];
  const float* b_hh = (const float*)d_in[4];
  const int*   nsz  = (const int*)d_in[5];
  float* out = (float*)d_out;

  // workspace layout (bytes), all offsets 256B-aligned; total ~258 MB
  char* ws = (char*)d_ws;
  float*  pf32   = (float*)(ws + 0);           // 33,554,432  padded f32
  __bf16* pb16   = (__bf16*)(ws + 33554432);   // 16,777,216  padded bf16
  __bf16* wihb   = (__bf16*)(ws + 50331648);   //  3,145,728  w_ih bf16
  __bf16* whhb   = (__bf16*)(ws + 53477376);   //  3,145,728  w_hh bf16
  float*  h0     = (float*)(ws + 56623104);    //    524,288  h0
  int*    starts = (int*)(ws + 57147392);      //      1,024  prefix sums
  float*  xw     = (float*)(ws + 57148416);    // 201,326,592 xw f32

  k_zero   <<<8192, 256, 0, stream>>>((u32x4*)pf32, (u32x4*)pb16);
  k_starts <<<1, 256, 0, stream>>>(nsz, starts);
  k_cvt_w  <<<6144, 256, 0, stream>>>(w_ih, w_hh, wihb, whhb);
  k_scatter<<<16384, 128, 0, stream>>>(atom, nsz, starts, pf32, pb16);
  k_h0     <<<512, 256, 0, stream>>>(pf32, h0);
  k_xproj  <<<24576, 256, 0, stream>>>(pb16, wihb, b_ih, xw);
  k_gru    <<<32, 256, 0, stream>>>(xw, h0, whhb, b_hh, nsz, out);
}
